// CausalSelfAttention_57690000720193
// MI455X (gfx1250) — compile-verified, matrix operand order checked by reading, not run on hardware
//
#include <hip/hip_runtime.h>

typedef __attribute__((ext_vector_type(16))) __bf16 v16bf;
typedef __attribute__((ext_vector_type(8)))  float  v8f;
typedef __attribute__((ext_vector_type(4)))  float  v4f;
typedef __attribute__((ext_vector_type(4)))  __bf16 v4bf;
typedef __attribute__((ext_vector_type(4)))  unsigned int v4u;

#define B_   2
#define T_   2048
#define C_   1024
#define H_   16
#define HD_  64
#define N3C_ 3072
#define M_   (B_ * T_)   // 4096

__device__ __forceinline__ v8f wmma_bf16(v16bf a, v16bf b, v8f c) {
    return __builtin_amdgcn_wmma_f32_16x16x32_bf16(
        /*neg_a=*/false, a, /*neg_b=*/false, b,
        /*c_mod=*/(short)0, c, /*reuse_a=*/false, /*reuse_b=*/false);
}

// Load a 16-bit WMMA A/B fragment whose contraction dim is contiguous at p[0..31].
// Per lane: elements 0-7 = K[8*half .. 8*half+7], elements 8-15 = K[16+8*half ..].
// p must be 16-byte aligned. Lowers to 2x b128 loads (ds or global).
__device__ __forceinline__ v16bf load_frag32(const __bf16* p, int half) {
    const v4u* q = (const v4u*)p;
    union { v4u u[2]; v16bf b; } t;
    t.u[0] = q[half];
    t.u[1] = q[2 + half];
    return t.b;
}

// ---------------------------------------------------------------------------
// Kernel 1: qkv = x @ w_qkv + b_qkv, scattered to Q/K/V [B,H,T,hd] in bf16
// ---------------------------------------------------------------------------
__global__ void __launch_bounds__(256)
qkv_proj_kernel(const float* __restrict__ x, const float* __restrict__ w,
                const float* __restrict__ bias,
                __bf16* __restrict__ Qo, __bf16* __restrict__ Ko, __bf16* __restrict__ Vo)
{
    __shared__ __attribute__((aligned(16))) __bf16 As[128][40];   // [row][k]
    __shared__ __attribute__((aligned(16))) __bf16 Bt[128][40];   // [col][k] (transposed)
    const int m0 = blockIdx.x * 128;
    const int n0 = blockIdx.y * 128;
    const int tid = threadIdx.x;
    const int lane = tid & 31, wave = tid >> 5;
    const int half = lane >> 4, r = lane & 15;
    const int wm = wave & 3, wn = wave >> 2;   // 4x2 wave grid -> 32x64 per wave

    v8f acc[2][4] = {};

    for (int k0 = 0; k0 < C_; k0 += 32) {
        // Stage A: float4 coalesced reads -> packed bf16 b64 LDS stores
        #pragma unroll
        for (int j = 0; j < 4; ++j) {
            int e = j * 1024 + tid * 4;            // element in 128x32 tile
            int mm = e >> 5, kk = e & 31;
            v4f f = *(const v4f*)&x[(size_t)(m0 + mm) * C_ + k0 + kk];
            v4bf b = { (__bf16)f[0], (__bf16)f[1], (__bf16)f[2], (__bf16)f[3] };
            *(v4bf*)&As[mm][kk] = b;
        }
        // Stage B transposed: coalesced f32 reads, scalar b16 transposed stores
        #pragma unroll
        for (int j = 0; j < 16; ++j) {
            int i = j * 256 + tid;                 // element in 32x128 tile
            int kk = i >> 7, nn = i & 127;
            Bt[nn][kk] = (__bf16)w[(size_t)(k0 + kk) * N3C_ + n0 + nn];
        }
        __syncthreads();

        v16bf afr[2], bfr[4];
        #pragma unroll
        for (int mi = 0; mi < 2; ++mi)
            afr[mi] = load_frag32(&As[wm * 32 + mi * 16 + r][0], half);
        #pragma unroll
        for (int ni = 0; ni < 4; ++ni)
            bfr[ni] = load_frag32(&Bt[wn * 64 + ni * 16 + r][0], half);

        #pragma unroll
        for (int mi = 0; mi < 2; ++mi)
            #pragma unroll
            for (int ni = 0; ni < 4; ++ni)
                acc[mi][ni] = wmma_bf16(afr[mi], bfr[ni], acc[mi][ni]);
        __syncthreads();
    }

    // Epilogue: this 128-wide N-tile lies entirely in one of Q/K/V
    const int which = n0 >> 10;
    __bf16* __restrict__ dst = (which == 0) ? Qo : (which == 1) ? Ko : Vo;
    for (int mi = 0; mi < 2; ++mi)
        for (int ni = 0; ni < 4; ++ni)
            #pragma unroll
            for (int v = 0; v < 8; ++v) {
                int m = m0 + wm * 32 + mi * 16 + v + 8 * half;
                int n = n0 + wn * 64 + ni * 16 + r;
                int b = m >> 11, t = m & (T_ - 1);
                int c = n & (C_ - 1);
                int h = c >> 6, d = c & 63;
                dst[(((size_t)b * H_ + h) * T_ + t) * HD_ + d] =
                    (__bf16)(acc[mi][ni][v] + bias[n]);
            }
}

// ---------------------------------------------------------------------------
// Kernel 2: fused causal flash attention. Block = 128 q-rows of one (b,h).
// 8 waves, 16 q-rows each; K/V tiles staged in LDS, shared by all waves.
// ---------------------------------------------------------------------------
__global__ void __launch_bounds__(256)
flash_attn_kernel(const __bf16* __restrict__ Q, const __bf16* __restrict__ K,
                  const __bf16* __restrict__ V, __bf16* __restrict__ Y)
{
    __shared__ __attribute__((aligned(16))) __bf16 Ks[32][72];       // [key][d]
    __shared__ __attribute__((aligned(16))) __bf16 Vst[64][40];      // [d][key] (transposed)
    __shared__ __attribute__((aligned(16))) __bf16 Ps[8][16][40];    // per-wave P tile

    const int bh = blockIdx.x;            // b*H + h
    const int q0 = blockIdx.y * 128;
    const int tid = threadIdx.x;
    const int lane = tid & 31, wave = tid >> 5;
    const int half = lane >> 4, r = lane & 15;
    const int qrow0 = q0 + wave * 16;

    const __bf16* Qh = Q + (size_t)bh * T_ * HD_;
    const __bf16* Kh = K + (size_t)bh * T_ * HD_;
    const __bf16* Vh = V + (size_t)bh * T_ * HD_;

    // Q A-fragments: 16 rows x hd=64 (two K=32 steps), 2x global b128 each
    v16bf qa[2];
    #pragma unroll
    for (int kh = 0; kh < 2; ++kh)
        qa[kh] = load_frag32(Qh + (size_t)(qrow0 + r) * HD_ + 32 * kh, half);

    v8f o[4] = {};
    float mrow[8], lrow[8];
    #pragma unroll
    for (int v = 0; v < 8; ++v) { mrow[v] = -1e30f; lrow[v] = 0.f; }
    const float scale = 0.125f;           // 1/sqrt(64)

    const int kend = q0 + 128;            // block-level causal key bound (<= T_)
    for (int kb = 0; kb < kend; kb += 32) {
        // Stage K (row-major, 8-byte copies) and V (transposed, scalar)
        #pragma unroll
        for (int j = 0; j < 2; ++j) {
            int ch = j * 256 + tid;              // 512 4-element chunks
            int kk = ch >> 4, dd = (ch & 15) * 4;
            *(v4bf*)&Ks[kk][dd] = *(const v4bf*)&Kh[(size_t)(kb + kk) * HD_ + dd];
        }
        #pragma unroll
        for (int j = 0; j < 8; ++j) {
            int i = j * 256 + tid;
            int kk = i >> 6, dd = i & 63;
            Vst[dd][kk] = Vh[(size_t)(kb + kk) * HD_ + dd];
        }
        __syncthreads();

        if (kb <= qrow0 + 15) {           // this wave still has unmasked keys
            // S = Q K^T for keys [kb, kb+32)
            v8f s0 = {}, s1 = {};
            #pragma unroll
            for (int kh = 0; kh < 2; ++kh) {
                v16bf kb0 = load_frag32(&Ks[r][32 * kh], half);
                v16bf kb1 = load_frag32(&Ks[16 + r][32 * kh], half);
                s0 = wmma_bf16(qa[kh], kb0, s0);
                s1 = wmma_bf16(qa[kh], kb1, s1);
            }
            // causal mask + online softmax (row = v + 8*half, col = r)
            float alpha[8];
            #pragma unroll
            for (int v = 0; v < 8; ++v) {
                int qg = qrow0 + v + 8 * half;
                float e0 = (kb + r      <= qg) ? s0[v] * scale : -1e30f;
                float e1 = (kb + 16 + r <= qg) ? s1[v] * scale : -1e30f;
                float mx = fmaxf(e0, e1);
                #pragma unroll
                for (int off = 1; off < 16; off <<= 1)
                    mx = fmaxf(mx, __shfl_xor(mx, off, 16));
                float mnew = fmaxf(mrow[v], mx);
                float a  = __expf(mrow[v] - mnew);
                float p0 = __expf(e0 - mnew);
                float p1 = __expf(e1 - mnew);
                float rs = p0 + p1;
                #pragma unroll
                for (int off = 1; off < 16; off <<= 1)
                    rs += __shfl_xor(rs, off, 16);
                mrow[v] = mnew;
                lrow[v] = lrow[v] * a + rs;
                alpha[v] = a;
                s0[v] = p0; s1[v] = p1;
            }
            #pragma unroll
            for (int ni = 0; ni < 4; ++ni)
                #pragma unroll
                for (int v = 0; v < 8; ++v) o[ni][v] *= alpha[v];

            // P (D-layout) -> LDS -> A-layout fragment (wave-private, DS in-order)
            #pragma unroll
            for (int v = 0; v < 8; ++v) {
                int row = v + 8 * half;
                Ps[wave][row][r]      = (__bf16)s0[v];
                Ps[wave][row][16 + r] = (__bf16)s1[v];
            }
            v16bf pa = load_frag32(&Ps[wave][r][0], half);

            // O += P V  (K=32 contraction over keys)
            #pragma unroll
            for (int ni = 0; ni < 4; ++ni) {
                v16bf vb = load_frag32(&Vst[ni * 16 + r][0], half);
                o[ni] = wmma_bf16(pa, vb, o[ni]);
            }
        }
        __syncthreads();
    }

    const int b = bh >> 4, h = bh & (H_ - 1);
    #pragma unroll
    for (int ni = 0; ni < 4; ++ni)
        #pragma unroll
        for (int v = 0; v < 8; ++v) {
            int t = qrow0 + v + 8 * half;
            float inv = lrow[v] > 0.f ? 1.f / lrow[v] : 0.f;
            Y[((size_t)b * T_ + t) * C_ + h * HD_ + ni * 16 + r] = (__bf16)(o[ni][v] * inv);
        }
}

// ---------------------------------------------------------------------------
// Kernel 3: out = y @ w_out + b_out  (fp32 output)
// ---------------------------------------------------------------------------
__global__ void __launch_bounds__(256)
out_proj_kernel(const __bf16* __restrict__ Yin, const float* __restrict__ w,
                const float* __restrict__ bias, float* __restrict__ out)
{
    __shared__ __attribute__((aligned(16))) __bf16 As[128][40];
    __shared__ __attribute__((aligned(16))) __bf16 Bt[128][40];
    const int m0 = blockIdx.x * 128;
    const int n0 = blockIdx.y * 128;
    const int tid = threadIdx.x;
    const int lane = tid & 31, wave = tid >> 5;
    const int half = lane >> 4, r = lane & 15;
    const int wm = wave & 3, wn = wave >> 2;

    v8f acc[2][4] = {};

    for (int k0 = 0; k0 < C_; k0 += 32) {
        // Stage A (already bf16): 8-byte copies
        #pragma unroll
        for (int j = 0; j < 4; ++j) {
            int ch = j * 256 + tid;              // 1024 4-element chunks
            int mm = ch >> 3, kk = (ch & 7) * 4;
            *(v4bf*)&As[mm][kk] = *(const v4bf*)&Yin[(size_t)(m0 + mm) * C_ + k0 + kk];
        }
        // Stage B transposed
        #pragma unroll
        for (int j = 0; j < 16; ++j) {
            int i = j * 256 + tid;
            int kk = i >> 7, nn = i & 127;
            Bt[nn][kk] = (__bf16)w[(size_t)(k0 + kk) * C_ + n0 + nn];
        }
        __syncthreads();

        v16bf afr[2], bfr[4];
        #pragma unroll
        for (int mi = 0; mi < 2; ++mi)
            afr[mi] = load_frag32(&As[wm * 32 + mi * 16 + r][0], half);
        #pragma unroll
        for (int ni = 0; ni < 4; ++ni)
            bfr[ni] = load_frag32(&Bt[wn * 64 + ni * 16 + r][0], half);

        #pragma unroll
        for (int mi = 0; mi < 2; ++mi)
            #pragma unroll
            for (int ni = 0; ni < 4; ++ni)
                acc[mi][ni] = wmma_bf16(afr[mi], bfr[ni], acc[mi][ni]);
        __syncthreads();
    }

    for (int mi = 0; mi < 2; ++mi)
        for (int ni = 0; ni < 4; ++ni)
            #pragma unroll
            for (int v = 0; v < 8; ++v) {
                int m = m0 + wm * 32 + mi * 16 + v + 8 * half;
                int n = n0 + wn * 64 + ni * 16 + r;
                out[(size_t)m * C_ + n] = acc[mi][ni][v] + bias[n];
            }
}

// ---------------------------------------------------------------------------
extern "C" void kernel_launch(void* const* d_in, const int* in_sizes, int n_in,
                              void* d_out, int out_size, void* d_ws, size_t ws_size,
                              hipStream_t stream)
{
    (void)in_sizes; (void)n_in; (void)out_size; (void)ws_size;
    const float* x     = (const float*)d_in[0];
    const float* w_qkv = (const float*)d_in[1];
    const float* b_qkv = (const float*)d_in[2];
    const float* w_out = (const float*)d_in[3];
    const float* b_out = (const float*)d_in[4];
    float* out = (float*)d_out;

    const size_t per = (size_t)B_ * H_ * T_ * HD_;   // 4,194,304 elements
    __bf16* Q = (__bf16*)d_ws;
    __bf16* K = Q + per;
    __bf16* V = K + per;
    __bf16* Y = V + per;                             // [B*T, C] bf16

    qkv_proj_kernel<<<dim3(M_ / 128, N3C_ / 128), 256, 0, stream>>>(x, w_qkv, b_qkv, Q, K, V);
    flash_attn_kernel<<<dim3(B_ * H_, T_ / 128), 256, 0, stream>>>(Q, K, V, Y);
    out_proj_kernel<<<dim3(M_ / 128, C_ / 128), 256, 0, stream>>>(Y, w_out, b_out, out);
}
